// MessagePassingGCN_3805341024624
// MI455X (gfx1250) — matrix operand order repaired and checked by reading, not compiled, not run on hardware
//
#include <hip/hip_runtime.h>

// ---------------------------------------------------------------------------
// 3-layer GCN for MI455X (gfx1250, wave32).
//   - Dense GEMMs on v_wmma_f32_16x16x32_bf16 (bf16 in, f32 accumulate).
//   - Edge aggregation: one wave32 per edge, coalesced 512B row gather,
//     4x global_atomic_add_f32 per lane into L2-resident accumulator.
//   - H (25.6MB) + accumulator (25.6MB) fit in the 192MB L2 -> atomics/gathers
//     stay on-chip; HBM floor ~90us for ~2GB of edge traffic.
// ---------------------------------------------------------------------------

#define HID 128

typedef __attribute__((ext_vector_type(16))) __bf16 v16bf;
typedef __attribute__((ext_vector_type(8)))  float  v8f;

union BfTile { v16bf v; uint4 q[2]; };

__device__ __forceinline__ unsigned short f2bf(float f) {
  union { float f; unsigned u; } c; c.f = f;
  unsigned u = c.u;
  unsigned r = u + 0x7fffu + ((u >> 16) & 1u);   // round-to-nearest-even
  return (unsigned short)(r >> 16);
}

// ---- elementwise helpers ---------------------------------------------------

__global__ void k_to_bf16(const float* __restrict__ in,
                          unsigned short* __restrict__ out, int n) {
  int i = blockIdx.x * blockDim.x + threadIdx.x;
  if (i < n) out[i] = f2bf(in[i]);
}

__global__ void k_fill(float* __restrict__ p, float v, int n) {
  int i = blockIdx.x * blockDim.x + threadIdx.x;
  if (i < n) p[i] = v;
}

__global__ void k_deg_accum(const int* __restrict__ dst,
                            float* __restrict__ deg, int ne) {
  int e = blockIdx.x * blockDim.x + threadIdx.x;
  if (e < ne) atomicAdd(&deg[dst[e]], 1.0f);
}

__global__ void k_rsqrt_inplace(float* __restrict__ d, int n) {
  int i = blockIdx.x * blockDim.x + threadIdx.x;
  if (i < n) d[i] = rsqrtf(d[i]);             // deg >= 1 (self loop)
}

// ---- WMMA GEMM: H[nrows,128] = A[nrows,128] * W^T, bf16 x bf16 -> f32 ------
// One wave per 16x16 output tile; K=128 as 4 steps of v_wmma_f32_16x16x32_bf16.

__global__ __launch_bounds__(256) void k_gemm_wmma(
    const unsigned short* __restrict__ A,   // [nrows,128] bf16, row major
    const unsigned short* __restrict__ Bt,  // [128,128]  bf16 = W row major (=B^T)
    float* __restrict__ Hf,                 // [nrows,128] f32
    int nrows, int ntiles)
{
  int wave = threadIdx.x >> 5;
  int lane = threadIdx.x & 31;
  int tile = blockIdx.x * 8 + wave;
  if (tile >= ntiles) return;
  const int ctiles = HID / 16;              // 8 column tiles
  int mtile = tile / ctiles;
  int ntile = tile % ctiles;

  int l16 = lane & 15;
  int lh  = lane >> 4;

  int mrow  = mtile * 16 + l16;
  int mload = mrow < nrows ? mrow : nrows - 1;     // clamp tail loads
  const unsigned short* arow = A  + (size_t)mload * HID;
  const unsigned short* brow = Bt + (size_t)(ntile * 16 + l16) * HID;

  v8f acc = {0.f, 0.f, 0.f, 0.f, 0.f, 0.f, 0.f, 0.f};
#pragma unroll
  for (int ks = 0; ks < 4; ++ks) {
    BfTile a, b;
    // A 16x32 bf16 layout: halfwords 0-7 -> K = 8*lh .. 8*lh+7,
    //                      halfwords 8-15 -> +16 (ISA 7.12.2)
    int kA = ks * 32 + lh * 8;
    a.q[0] = *(const uint4*)(arow + kA);
    a.q[1] = *(const uint4*)(arow + kA + 16);
    // B 32x16 bf16 layout: same K order as A (8*lh..8*lh+7, then +16)
    int kB = ks * 32 + lh * 8;
    b.q[0] = *(const uint4*)(brow + kB);
    b.q[1] = *(const uint4*)(brow + kB + 16);
    acc = __builtin_amdgcn_wmma_f32_16x16x32_bf16(
        /*neg_a=*/false, a.v, /*neg_b=*/false, b.v,
        /*c_mod=*/(short)0, acc, /*reuse_a=*/false, /*reuse_b=*/false);
  }

  // C/D f32 16x16: VGPR v -> M = 8*lh + v, N = l16
  // Store base: row (mtile*16 + 8*lh), column (ntile*16 + l16); rows stride HID.
  float* out = Hf + (size_t)(mtile * 16 + 8 * lh) * HID + ntile * 16 + l16;
  if (mtile * 16 + 16 <= nrows) {
    // Full tile (always true when nrows % 16 == 0): branchless coalesced stores.
#pragma unroll
    for (int v = 0; v < 8; ++v)
      out[(size_t)v * HID] = acc[v];
  } else {
    int mbase = mtile * 16 + 8 * lh;
#pragma unroll
    for (int v = 0; v < 8; ++v)
      if (mbase + v < nrows)
        out[(size_t)v * HID] = acc[v];
  }
}

// ---- edge scatter: out[dst] += dinv[src]*dinv[dst] * H[src] ----------------
// One wave32 per edge: 512B coalesced gather, 4 f32 atomics per lane.

__global__ __launch_bounds__(256) void k_edge_agg(
    const int* __restrict__ src, const int* __restrict__ dst,
    const float* __restrict__ dinv, const float* __restrict__ Hf,
    float* __restrict__ out, int ne)
{
  int tid  = blockIdx.x * blockDim.x + threadIdx.x;
  int e    = tid >> 5;
  if (e >= ne) return;
  int lane = tid & 31;
  int s = src[e], d = dst[e];
  float norm = dinv[s] * dinv[d];
  const float4 hv = *(const float4*)(Hf + (size_t)s * HID + lane * 4);
  float* o = out + (size_t)d * HID + lane * 4;
  atomicAdd(o + 0, norm * hv.x);
  atomicAdd(o + 1, norm * hv.y);
  atomicAdd(o + 2, norm * hv.z);
  atomicAdd(o + 3, norm * hv.w);
}

// ---- finalize: add self-loop term + bias; ReLU->bf16 or plain f32 ----------

__global__ void k_finalize(const float* __restrict__ agg,
                           const float* __restrict__ Hf,
                           const float* __restrict__ dinv,
                           const float* __restrict__ bias,
                           unsigned short* __restrict__ nextA, // bf16 out (layers 0,1)
                           float* __restrict__ fout,           // f32 out (layer 2)
                           int n)
{
  int i = blockIdx.x * blockDim.x + threadIdx.x;
  if (i >= n) return;
  int row = i >> 7;            // /128
  int col = i & (HID - 1);
  float di = dinv[row];
  float v = agg[i] + di * di * Hf[i] + bias[col];
  if (nextA) nextA[i] = f2bf(fmaxf(v, 0.0f));
  else       fout[i]  = v;
}

// ---------------------------------------------------------------------------

extern "C" void kernel_launch(void* const* d_in, const int* in_sizes, int n_in,
                              void* d_out, int out_size, void* d_ws, size_t ws_size,
                              hipStream_t stream) {
  const float* x  = (const float*)d_in[0];
  const int*   ei = (const int*)d_in[1];          // edge_index [2,E] int
  const float* W[3] = {(const float*)d_in[2], (const float*)d_in[4], (const float*)d_in[6]};
  const float* b[3] = {(const float*)d_in[3], (const float*)d_in[5], (const float*)d_in[7]};

  const int nfeat = in_sizes[0];                  // N*128
  const int nN    = nfeat / HID;                  // 50000
  const int nE    = in_sizes[1] / 2;              // 600000
  const int* src  = ei;
  const int* dst  = ei + nE;

  // workspace carve-out (256B aligned)
  char*  ws  = (char*)d_ws;
  size_t off = 0;
  auto alloc = [&](size_t bytes) -> char* {
    char* p = ws + off;
    off += (bytes + 255) & ~(size_t)255;
    return p;
  };
  float*          dinv = (float*)         alloc((size_t)nN * 4);
  unsigned short* Abf  = (unsigned short*)alloc((size_t)nfeat * 2);
  unsigned short* Wbf  = (unsigned short*)alloc((size_t)HID * HID * 2);
  float*          Hbuf = (float*)         alloc((size_t)nfeat * 4);
  float*          agg  = (float*)d_out;           // accumulator doubles as output

  const int T = 256;
  auto nb = [&](long long n) -> int { return (int)((n + T - 1) / T); };

  // degrees: deg = 1 (self loop) + sum over edges; dinv = rsqrt(deg)
  k_fill<<<nb(nN), T, 0, stream>>>(dinv, 1.0f, nN);
  k_deg_accum<<<nb(nE), T, 0, stream>>>(dst, dinv, nE);
  k_rsqrt_inplace<<<nb(nN), T, 0, stream>>>(dinv, nN);

  // layer input -> bf16
  k_to_bf16<<<nb(nfeat), T, 0, stream>>>(x, Abf, nfeat);

  const int ntiles = ((nN + 15) / 16) * (HID / 16);
  for (int l = 0; l < 3; ++l) {
    k_to_bf16<<<nb(HID * HID), T, 0, stream>>>(W[l], Wbf, HID * HID);
    k_gemm_wmma<<<(ntiles + 7) / 8, 256, 0, stream>>>(Abf, Wbf, Hbuf, nN, ntiles);
    k_fill<<<nb(nfeat), T, 0, stream>>>(agg, 0.0f, nfeat);
    k_edge_agg<<<nb((long long)nE * 32), T, 0, stream>>>(src, dst, dinv, Hbuf, agg, nE);
    const bool last = (l == 2);
    k_finalize<<<nb(nfeat), T, 0, stream>>>(agg, Hbuf, dinv, b[l],
                                            last ? nullptr : Abf,
                                            last ? agg : nullptr, nfeat);
  }
}
